// PositionalAttentionPooling_47794396070381
// MI455X (gfx1250) — matrix operand order repaired and checked by reading, not compiled, not run on hardware
//
#include <hip/hip_runtime.h>

// ---------------------------------------------------------------------------
// Types for CDNA5 WMMA (wave32): v_wmma_f32_16x16x32_bf16
// ---------------------------------------------------------------------------
typedef __attribute__((ext_vector_type(16))) __bf16 v16bf;
typedef __attribute__((ext_vector_type(8)))  float  v8f;

#define DLR 78      // real DL
#define DLP 80      // padded DL (5 x 16)
#define MSLC 50
#define DC 128

__device__ __forceinline__ __bf16 f2bf(float x) { return (__bf16)x; }

// ---------------------------------------------------------------------------
// Shared WMMA micro-kernel: one wave computes a 16 x (NT*16) tile of
// A[16x128] (bf16, row-major in LDS) @ W^T where W is [NT*16 x 128] bf16 LDS.
// Fragment layouts per CDNA5 ISA 7.12.2:
//  A (16-bit, 16x32): lanes 0-15 -> M=0..15, K in {h*8+0..7, 16+h*8+0..7}
//  B (16-bit, 32x16): lane -> N=lane%16, elements K = h*8 + e (e<8), 16 + h*8 + e-8 (e>=8)
//  C/D (f32 16x16):   lane%16 -> N, vgpr v -> M = v + 8*(lane/16)
// ---------------------------------------------------------------------------
template <int NT>
__device__ __forceinline__ void wmma_rowtile(const __bf16* A /*16x128*/,
                                             const __bf16* W /*(NT*16)x128*/,
                                             v8f acc[NT]) {
  const int lane = threadIdx.x & 31;
  const int mr   = lane & 15;
  const int hf   = lane >> 4;
#pragma unroll
  for (int kb = 0; kb < 4; ++kb) {            // K = 128 in 4 steps of 32
    v16bf af;
#pragma unroll
    for (int j = 0; j < 8; ++j) {
      int kk = kb * 32 + ((j >= 4) ? 16 : 0) + hf * 8 + 2 * (j & 3);
      af[2 * j]     = A[mr * 128 + kk];
      af[2 * j + 1] = A[mr * 128 + kk + 1];
    }
#pragma unroll
    for (int nt = 0; nt < NT; ++nt) {
      v16bf bfrag;
      const int n   = nt * 16 + mr;
      const int kb0 = kb * 32 + hf * 8;
#pragma unroll
      for (int e = 0; e < 16; ++e) bfrag[e] = W[n * 128 + kb0 + e + ((e >= 8) ? 8 : 0)];
      acc[nt] = __builtin_amdgcn_wmma_f32_16x16x32_bf16(
          false, af, false, bfrag, (short)0, acc[nt], false, false);
    }
  }
}

// ---------------------------------------------------------------------------
// Prefix scan of cnt[] (block sums -> exclusive scan -> scatter src_map)
// ---------------------------------------------------------------------------
__global__ void k_block_sums(const int* __restrict__ cnt, int* __restrict__ bsum, int P) {
  __shared__ int s[256];
  int p = blockIdx.x * 256 + threadIdx.x;
  s[threadIdx.x] = (p < P) ? cnt[p] : 0;
  __syncthreads();
  for (int off = 128; off > 0; off >>= 1) {
    if (threadIdx.x < off) s[threadIdx.x] += s[threadIdx.x + off];
    __syncthreads();
  }
  if (threadIdx.x == 0) bsum[blockIdx.x] = s[0];
}

__global__ void k_scan_single(int* bsum, int nb) {
  if (blockIdx.x == 0 && threadIdx.x == 0) {
    int run = 0;
    for (int i = 0; i < nb; ++i) { int v = bsum[i]; bsum[i] = run; run += v; }
  }
}

__global__ void k_scatter_src(const int* __restrict__ cnt, const int* __restrict__ bsum,
                              int* __restrict__ src_map, int P) {
  __shared__ int s[256];
  int tid = threadIdx.x;
  int p = blockIdx.x * 256 + tid;
  int c = (p < P) ? cnt[p] : 0;
  s[tid] = c;
  __syncthreads();
  for (int off = 1; off < 256; off <<= 1) {   // inclusive Hillis-Steele
    int v = (tid >= off) ? s[tid - off] : 0;
    __syncthreads();
    s[tid] += v;
    __syncthreads();
  }
  int start = bsum[blockIdx.x] + s[tid] - c;
  for (int k = 0; k < c; ++k) src_map[start + k] = p;
}

// ---------------------------------------------------------------------------
// Query branch: qe = tanh([qfeat @ Wq^T + bq ; pos]) -> node_emb rows T..T+NQ
// 4 waves/block, 64 rows/block, NT=5 (80 padded cols, keep 78)
// ---------------------------------------------------------------------------
__global__ void k_qe(const float* __restrict__ qfeat, const float* __restrict__ Wq,
                     const float* __restrict__ bq, const float* __restrict__ pos_table,
                     const int* __restrict__ qpos, const int* __restrict__ qbatch,
                     float* __restrict__ node_emb, int* __restrict__ node_batch,
                     int NQ, int T) {
  __shared__ __bf16 Wlds[DLP * 128];
  __shared__ __bf16 Alds[64 * 128];
  __shared__ float  blds[DLP];
  const int tid = threadIdx.x;
  for (int i = tid; i < DLP * 128; i += 128) {
    int n = i >> 7, k = i & 127;
    Wlds[i] = f2bf((n < DLR) ? Wq[n * 128 + k] : 0.f);
  }
  if (tid < DLP) blds[tid] = (tid < DLR) ? bq[tid] : 0.f;
  const int row0 = blockIdx.x * 64;
  for (int i = tid; i < 64 * 128; i += 128) {
    int r = i >> 7;
    int gr = row0 + r; if (gr >= NQ) gr = NQ - 1;
    Alds[i] = f2bf(qfeat[(long)gr * 128 + (i & 127)]);
  }
  __syncthreads();
  const int wv = tid >> 5, lane = tid & 31;
  const int col = lane & 15, hf = lane >> 4;
  v8f acc[5] = {};
  wmma_rowtile<5>(&Alds[wv * 16 * 128], Wlds, acc);
#pragma unroll
  for (int nt = 0; nt < 5; ++nt) {
#pragma unroll
    for (int v = 0; v < 8; ++v) {
      int grow = row0 + wv * 16 + v + 8 * hf;
      int n = nt * 16 + col;
      if (grow < NQ && n < DLR)
        node_emb[(long)(T + grow) * 128 + n] = tanhf(acc[nt][v] + blds[n]);
    }
  }
  for (int i = tid; i < 64 * MSLC; i += 128) {
    int r = i / MSLC, d = i % MSLC;
    int grow = row0 + r;
    if (grow < NQ)
      node_emb[(long)(T + grow) * 128 + DLR + d] = tanhf(pos_table[qpos[grow] * MSLC + d]);
  }
  for (int i = tid; i < 64; i += 128) {
    int grow = row0 + i;
    if (grow < NQ) node_batch[T + grow] = qbatch[grow];
  }
}

// ---------------------------------------------------------------------------
// Product linear at P granularity: peLin[p][0..79] = pfeat @ Wp^T + bp
// ---------------------------------------------------------------------------
__global__ void k_pe(const float* __restrict__ pfeat, const float* __restrict__ Wp,
                     const float* __restrict__ bp, float* __restrict__ peLin, int P) {
  __shared__ __bf16 Wlds[DLP * 128];
  __shared__ __bf16 Alds[64 * 128];
  __shared__ float  blds[DLP];
  const int tid = threadIdx.x;
  for (int i = tid; i < DLP * 128; i += 128) {
    int n = i >> 7, k = i & 127;
    Wlds[i] = f2bf((n < DLR) ? Wp[n * 128 + k] : 0.f);
  }
  if (tid < DLP) blds[tid] = (tid < DLR) ? bp[tid] : 0.f;
  const int row0 = blockIdx.x * 64;
  for (int i = tid; i < 64 * 128; i += 128) {
    int r = i >> 7;
    int gr = row0 + r; if (gr >= P) gr = P - 1;
    Alds[i] = f2bf(pfeat[(long)gr * 128 + (i & 127)]);
  }
  __syncthreads();
  const int wv = tid >> 5, lane = tid & 31;
  const int col = lane & 15, hf = lane >> 4;
  v8f acc[5] = {};
  wmma_rowtile<5>(&Alds[wv * 16 * 128], Wlds, acc);
#pragma unroll
  for (int nt = 0; nt < 5; ++nt) {
#pragma unroll
    for (int v = 0; v < 8; ++v) {
      int grow = row0 + wv * 16 + v + 8 * hf;
      int n = nt * 16 + col;
      if (grow < P) peLin[(long)grow * DLP + n] = acc[nt][v] + blds[n];
    }
  }
}

// ---------------------------------------------------------------------------
// Expand products by cnt (via src_map), concat pos emb, tanh -> node_emb[0..T)
// ---------------------------------------------------------------------------
__global__ void k_expand(const float* __restrict__ peLin, const int* __restrict__ src_map,
                         const int* __restrict__ ppos, const int* __restrict__ pbatch,
                         const float* __restrict__ pos_table,
                         float* __restrict__ node_emb, int* __restrict__ node_batch, int T) {
  int idx = blockIdx.x * 256 + threadIdx.x;
  if (idx >= T * 128) return;
  int t = idx >> 7, d = idx & 127;
  int sp = src_map[t];
  float v = (d < DLR) ? peLin[(long)sp * DLP + d]
                      : pos_table[ppos[t] * MSLC + (d - DLR)];
  node_emb[idx] = tanhf(v);
  if (d == 0) node_batch[t] = pbatch[sp];
}

// ---------------------------------------------------------------------------
// Segment sums (L2 f32 atomics)
// ---------------------------------------------------------------------------
__global__ void k_zero(float* p, int n) {
  int i = blockIdx.x * 256 + threadIdx.x;
  if (i < n) p[i] = 0.f;
}

__global__ void k_accum_coarse(const float* __restrict__ node_emb, const int* __restrict__ node_batch,
                               float* __restrict__ coarse, float* __restrict__ counts, int N) {
  int idx = blockIdx.x * 256 + threadIdx.x;
  if (idx >= N * 128) return;
  int i = idx >> 7, d = idx & 127;
  int b = node_batch[i];
  atomicAdd(&coarse[b * 128 + d], node_emb[idx]);
  if (d == 0) atomicAdd(&counts[b], 1.0f);
}

// ---------------------------------------------------------------------------
// cproj[b] = (coarse[b]/cnt[b]) @ Wc^T + bn   (B x 128 GEMM, NT=8)
// ---------------------------------------------------------------------------
__global__ void k_cproj(const float* __restrict__ coarse, const float* __restrict__ counts,
                        const float* __restrict__ Wc, const float* __restrict__ bn,
                        float* __restrict__ cproj, int B) {
  __shared__ __bf16 Wlds[128 * 128];
  __shared__ __bf16 Alds[64 * 128];
  const int tid = threadIdx.x;
  for (int i = tid; i < 128 * 128; i += 128) Wlds[i] = f2bf(Wc[i]);
  const int row0 = blockIdx.x * 64;
  for (int i = tid; i < 64 * 128; i += 128) {
    int r = i >> 7;
    int gr = row0 + r; if (gr >= B) gr = B - 1;
    float inv = 1.f / fmaxf(counts[gr], 1.f);
    Alds[i] = f2bf(coarse[(long)gr * 128 + (i & 127)] * inv);
  }
  __syncthreads();
  const int wv = tid >> 5, lane = tid & 31;
  const int col = lane & 15, hf = lane >> 4;
  v8f acc[8] = {};
  wmma_rowtile<8>(&Alds[wv * 16 * 128], Wlds, acc);
#pragma unroll
  for (int nt = 0; nt < 8; ++nt) {
#pragma unroll
    for (int v = 0; v < 8; ++v) {
      int grow = row0 + wv * 16 + v + 8 * hf;
      if (grow < B) cproj[(long)grow * 128 + nt * 16 + col] = acc[nt][v] + bn[nt * 16 + col];
    }
  }
}

// ---------------------------------------------------------------------------
// att[i] = sum_n sigmoid( (node_emb @ Wn^T)[i,n] + cproj[batch[i],n] ) * Wa[n]
// Fused WMMA GEMM + epilogue; cross-lane row reduction via LDS using the
// C-matrix lane/VGPR layout.
// ---------------------------------------------------------------------------
__global__ void k_att(const float* __restrict__ node_emb, const int* __restrict__ node_batch,
                      const float* __restrict__ Wn, const float* __restrict__ cproj,
                      const float* __restrict__ Wa, float* __restrict__ att, int N) {
  __shared__ __bf16 Wlds[128 * 128];
  __shared__ __bf16 Alds[64 * 128];
  __shared__ float  red[4][16][16];
  const int tid = threadIdx.x;
  for (int i = tid; i < 128 * 128; i += 128) Wlds[i] = f2bf(Wn[i]);
  const int row0 = blockIdx.x * 64;
  for (int i = tid; i < 64 * 128; i += 128) {
    int r = i >> 7;
    int gr = row0 + r; if (gr >= N) gr = N - 1;
    Alds[i] = f2bf(node_emb[(long)gr * 128 + (i & 127)]);
  }
  __syncthreads();
  const int wv = tid >> 5, lane = tid & 31;
  const int col = lane & 15, hf = lane >> 4;
  v8f acc[8] = {};
  wmma_rowtile<8>(&Alds[wv * 16 * 128], Wlds, acc);
  float wa[8];
#pragma unroll
  for (int nt = 0; nt < 8; ++nt) wa[nt] = Wa[nt * 16 + col];
#pragma unroll
  for (int v = 0; v < 8; ++v) {
    int mloc = v + 8 * hf;
    int grow = row0 + wv * 16 + mloc;
    float partial = 0.f;
    if (grow < N) {
      int b = node_batch[grow];
#pragma unroll
      for (int nt = 0; nt < 8; ++nt) {
        float z = acc[nt][v] + cproj[(long)b * 128 + nt * 16 + col];
        partial += wa[nt] / (1.f + __expf(-z));
      }
    }
    red[wv][mloc][col] = partial;
  }
  __syncthreads();
  if (lane < 16) {
    int grow = row0 + wv * 16 + lane;
    if (grow < N) {
      float s = 0.f;
#pragma unroll
      for (int c2 = 0; c2 < 16; ++c2) s += red[wv][lane][c2];
      att[grow] = s;
    }
  }
}

__global__ void k_accum_weighted(const float* __restrict__ node_emb, const int* __restrict__ node_batch,
                                 const float* __restrict__ att, float* __restrict__ wsum, int N) {
  int idx = blockIdx.x * 256 + threadIdx.x;
  if (idx >= N * 128) return;
  int i = idx >> 7, d = idx & 127;
  atomicAdd(&wsum[node_batch[i] * 128 + d], node_emb[idx] * att[i]);
}

__global__ void k_finalize(const float* __restrict__ wsum, const float* __restrict__ counts,
                           float* __restrict__ out, int B) {
  int idx = blockIdx.x * 256 + threadIdx.x;
  if (idx >= B * 128) return;
  out[idx] = wsum[idx] / fmaxf(counts[idx >> 7], 1.f);
}

// ---------------------------------------------------------------------------
// Host orchestration
// ---------------------------------------------------------------------------
extern "C" void kernel_launch(void* const* d_in, const int* in_sizes, int n_in,
                              void* d_out, int out_size, void* d_ws, size_t ws_size,
                              hipStream_t stream) {
  const float* query_feat   = (const float*)d_in[0];
  const float* product_feat = (const float*)d_in[1];
  const float* Wq           = (const float*)d_in[2];
  const float* bq           = (const float*)d_in[3];
  const float* Wp           = (const float*)d_in[4];
  const float* bp           = (const float*)d_in[5];
  const float* pos_table    = (const float*)d_in[6];
  const float* Wn           = (const float*)d_in[7];
  const float* bn           = (const float*)d_in[8];
  const float* Wc           = (const float*)d_in[9];
  const float* Wa           = (const float*)d_in[10];
  const int*   qpos         = (const int*)d_in[11];
  const int*   ppos         = (const int*)d_in[12];
  const int*   qbatch       = (const int*)d_in[13];
  const int*   pbatch       = (const int*)d_in[14];
  const int*   cnt          = (const int*)d_in[15];

  const int NQ = in_sizes[11];
  const int T  = in_sizes[12];
  const int P  = in_sizes[15];
  const int B  = out_size / 128;
  const int N  = T + NQ;

  // workspace carve-out (256B aligned)
  char*  ws  = (char*)d_ws;
  size_t off = 0;
  auto carve = [&](size_t bytes) -> void* {
    void* p = ws + off;
    off += (bytes + 255) & ~(size_t)255;
    return p;
  };
  const int nbP = (P + 255) / 256;
  float* peLin     = (float*)carve((size_t)P * DLP * sizeof(float));
  int*   src_map   = (int*)carve((size_t)T * sizeof(int));
  int*   bsum      = (int*)carve((size_t)nbP * sizeof(int));
  int*   node_bat  = (int*)carve((size_t)N * sizeof(int));
  float* node_emb  = (float*)carve((size_t)N * 128 * sizeof(float));
  float* coarse    = (float*)carve((size_t)B * 128 * sizeof(float));
  float* counts    = (float*)carve((size_t)B * sizeof(float));
  float* cproj     = (float*)carve((size_t)B * 128 * sizeof(float));
  float* attv      = (float*)carve((size_t)N * sizeof(float));
  float* wsum      = (float*)carve((size_t)B * 128 * sizeof(float));
  (void)ws_size; (void)n_in;

  // zero accumulators (re-zeroed every call: deterministic work)
  k_zero<<<(B * 128 + 255) / 256, 256, 0, stream>>>(coarse, B * 128);
  k_zero<<<(B + 255) / 256, 256, 0, stream>>>(counts, B);
  k_zero<<<(B * 128 + 255) / 256, 256, 0, stream>>>(wsum, B * 128);

  // repeat_interleave mapping via scan of cnt
  k_block_sums<<<nbP, 256, 0, stream>>>(cnt, bsum, P);
  k_scan_single<<<1, 32, 0, stream>>>(bsum, nbP);
  k_scatter_src<<<nbP, 256, 0, stream>>>(cnt, bsum, src_map, P);

  // branch GEMMs (bf16 WMMA, f32 accumulate)
  k_qe<<<(NQ + 63) / 64, 128, 0, stream>>>(query_feat, Wq, bq, pos_table, qpos, qbatch,
                                           node_emb, node_bat, NQ, T);
  k_pe<<<(P + 63) / 64, 128, 0, stream>>>(product_feat, Wp, bp, peLin, P);
  k_expand<<<((long)T * 128 + 255) / 256, 256, 0, stream>>>(peLin, src_map, ppos, pbatch,
                                                            pos_table, node_emb, node_bat, T);

  // coarse segment mean -> session projection (small WMMA GEMM)
  k_accum_coarse<<<((long)N * 128 + 255) / 256, 256, 0, stream>>>(node_emb, node_bat, coarse, counts, N);
  k_cproj<<<(B + 63) / 64, 128, 0, stream>>>(coarse, counts, Wc, bn, cproj, B);

  // attention (big WMMA GEMM fused with sigmoid/Wa reduction)
  k_att<<<(N + 63) / 64, 128, 0, stream>>>(node_emb, node_bat, Wn, cproj, Wa, attv, N);

  // weighted segment mean -> output
  k_accum_weighted<<<((long)N * 128 + 255) / 256, 256, 0, stream>>>(node_emb, node_bat, attv, wsum, N);
  k_finalize<<<(B * 128 + 255) / 256, 256, 0, stream>>>(wsum, counts, (float*)d_out, B);
}
